// Network_67851893342670
// MI455X (gfx1250) — compile-verified, matrix operand order checked by reading, not run on hardware
//
#include <hip/hip_runtime.h>
#include <hip/hip_bf16.h>

// ---------------------------------------------------------------------------
// Types
// ---------------------------------------------------------------------------
typedef _Float16 v16h __attribute__((ext_vector_type(16)));
typedef _Float16 v8h  __attribute__((ext_vector_type(8)));
typedef _Float16 v4h  __attribute__((ext_vector_type(4)));
typedef float    v8f  __attribute__((ext_vector_type(8)));

typedef unsigned int uint32x4 __attribute__((ext_vector_type(4)));
typedef int          int32x8  __attribute__((ext_vector_type(8)));
typedef int          int32x4  __attribute__((ext_vector_type(4)));

#define NB   2          // batch
#define KDIM 15232      // 17*(128*3*2) + 17*128
#define LSTR 40         // LDS row stride in halfs (32 data + 8 pad, keeps 16B align)

// ---------------------------------------------------------------------------
// Weight conversion kernels
// ---------------------------------------------------------------------------
// Pack OIHW f32 weights into [Cout][ky*kx][Cin] f16 so that K-chunks of the
// implicit GEMM (K ordered as (kpos, ci)) are contiguous 16B-aligned vectors.
__global__ void wpack_kernel(const float* __restrict__ w, _Float16* __restrict__ o,
                             int Cout, int Cin, int ks)
{
    size_t idx   = (size_t)blockIdx.x * 256 + threadIdx.x;
    size_t total = (size_t)Cout * Cin * ks * ks;
    if (idx >= total) return;
    int ks2 = ks * ks;
    int kx  = (int)(idx % ks);
    int ky  = (int)((idx / ks) % ks);
    int ci  = (int)((idx / ks2) % Cin);
    int co  = (int)(idx / ((size_t)ks2 * Cin));
    size_t oidx = (size_t)co * ks2 * Cin + (size_t)(ky * ks + kx) * Cin + ci;
    o[oidx] = (_Float16)w[idx];
}

// Straight f32 -> f16 copy (OIHW preserved) for direct-conv kernels.
__global__ void wplain_kernel(const float* __restrict__ w, _Float16* __restrict__ o,
                              size_t total)
{
    size_t idx = (size_t)blockIdx.x * 256 + threadIdx.x;
    if (idx < total) o[idx] = (_Float16)w[idx];
}

// ---------------------------------------------------------------------------
// WMMA implicit-GEMM conv (3x3 or 1x1, stride 1 or 2, NCHW f16 activations)
//   GEMM: M = Cout, N = flattened output pixels of one image, K = Cin*ks*ks
//   ordered (kpos, ci); requires Cin % 32 == 0 and Wo a power of two.
//   Block = 256 threads = 8 wave32s arranged WM x WN (WN = 8/WM); each wave
//   computes a 32x32 output tile as 2x2 WMMA 16x16 tiles (register blocking:
//   4 v_wmma per K-step, A/B fragments reused across the 2x2 product).
//   Software pipeline: chunk k+1 is gathered from global into registers while
//   the WMMAs for chunk k execute; registers commit to LDS at the next step.
//   WM=4 -> 128x64 block tile, WM=2 -> 64x128, WM=1 -> 32x256.
// ---------------------------------------------------------------------------
template<int WM>
__global__ __launch_bounds__(256)
void conv_wmma_kernel(const _Float16* __restrict__ X,
                      const _Float16* __restrict__ Wp,    // packed [Cout][ks2][Cin]
                      const float*    __restrict__ bias,
                      const _Float16* __restrict__ res,   // optional residual (same shape as Y)
                      _Float16*       __restrict__ Y,
                      int Cin, int Cout, int H, int W, int Ho, int Wo, int woShift,
                      int ks, int stride, int relu)
{
    constexpr int WN = 8 / WM;
    constexpr int MT = WM * 32;
    constexpr int NT = WN * 32;
    __shared__ _Float16 lA[MT * LSTR];   // [co_local][k_local] (+pad)
    __shared__ _Float16 lB[NT * LSTR];   // [px_local][k_local] (+pad)

    const int tid  = threadIdx.x;
    const int lane = tid & 31;
    const int wave = tid >> 5;
    const int wm   = wave % WM;
    const int wn   = wave / WM;

    const int n   = blockIdx.z;
    const int p0  = blockIdx.x * NT;     // flattened output-pixel base (oh*Wo+ow)
    const int co0 = blockIdx.y * MT;

    const int pad  = ks >> 1;
    const int K    = Cin * ks * ks;
    const int wrow = ks * ks * Cin;      // halfs per Cout row of packed weights
    const int wmask = Wo - 1;

    v8f acc[2][2] = {};

    // A staging: each thread owns AL contiguous K halfs of one Cout row.
    constexpr int AL = MT / 8;           // 16 / 8 / 4 halfs per thread
    const int a_co = tid / (32 / AL);
    const int a_k  = (tid % (32 / AL)) * AL;
    // B staging: each thread owns NT/8 pixels for one K element; adjacent
    // lanes handle adjacent pixels (coalesced within groups of 8).
    constexpr int PXT = NT / 8;
    const int b_k  = tid >> 3;
    const int b_pl = tid & 7;

    // ---- pipeline staging registers ----
    v8h      aR[AL >= 8 ? AL / 8 : 1];
    v4h      aR4;
    _Float16 bR[PXT];

    auto loadA = [&](int kk0) {
        const int kpos = kk0 / Cin;
        const int ci0  = kk0 - kpos * Cin;
        const _Float16* src =
            Wp + (size_t)(co0 + a_co) * wrow + (size_t)kpos * Cin + ci0 + a_k;
        if constexpr (AL >= 8) {
            #pragma unroll
            for (int c = 0; c < AL / 8; ++c) aR[c] = *(const v8h*)(src + 8 * c);
        } else {
            aR4 = *(const v4h*)src;
        }
    };
    auto loadB = [&](int kk0) {
        const int kpos = kk0 / Cin;
        const int ci0  = kk0 - kpos * Cin;
        const int ci = ci0 + b_k;
        const int ky = kpos / ks;
        const int kx = kpos - ky * ks;
        const _Float16* xpl = X + (size_t)(n * Cin + ci) * H * (size_t)W;
        #pragma unroll
        for (int r = 0; r < PXT; ++r) {
            const int p  = p0 + b_pl + 8 * r;
            const int oh = p >> woShift;
            const int ow = p & wmask;
            const int ih = oh * stride + ky - pad;
            const int iw = ow * stride + kx - pad;
            _Float16 v = (_Float16)0.f;
            if ((unsigned)ih < (unsigned)H && (unsigned)iw < (unsigned)W)
                v = xpl[(size_t)ih * W + iw];
            bR[r] = v;
        }
    };
    auto storeLDS = [&]() {
        if constexpr (AL >= 8) {
            #pragma unroll
            for (int c = 0; c < AL / 8; ++c)
                *(v8h*)&lA[a_co * LSTR + a_k + 8 * c] = aR[c];
        } else {
            *(v4h*)&lA[a_co * LSTR + a_k] = aR4;
        }
        #pragma unroll
        for (int r = 0; r < PXT; ++r)
            lB[(b_pl + 8 * r) * LSTR + b_k] = bR[r];
    };

    loadA(0);
    loadB(0);

    for (int kk0 = 0; kk0 < K; kk0 += 32) {
        storeLDS();
        __syncthreads();

        // Issue next chunk's global gathers; their wait lands at the next
        // storeLDS, fully overlapped with the WMMA sequence below.
        if (kk0 + 32 < K) {
            loadA(kk0 + 32);
            loadB(kk0 + 32);
        }
        if (kk0 + 64 < K) {
            const int kp2 = (kk0 + 64) / Cin;
            const int ci2 = (kk0 + 64) - kp2 * Cin;
            __builtin_prefetch(Wp + (size_t)(co0 + a_co) * wrow +
                               (size_t)kp2 * Cin + ci2 + a_k, 0, 1);
        }

        // ---- fragments per the CDNA5 16-bit A/B VGPR layout, 4 WMMAs ----
        {
            const int klo = (lane >= 16) ? 8 : 0;
            const int l15 = lane & 15;
            v16h Af[2], Bf[2];
            #pragma unroll
            for (int mi = 0; mi < 2; ++mi) {
                const int ar = wm * 32 + mi * 16 + l15;
                v8h a0 = *(const v8h*)&lA[ar * LSTR + klo];
                v8h a1 = *(const v8h*)&lA[ar * LSTR + klo + 16];
                #pragma unroll
                for (int i = 0; i < 8; ++i) { Af[mi][i] = a0[i]; Af[mi][i + 8] = a1[i]; }
            }
            #pragma unroll
            for (int ni = 0; ni < 2; ++ni) {
                const int bc = wn * 32 + ni * 16 + l15;
                v8h b0 = *(const v8h*)&lB[bc * LSTR + klo];
                v8h b1 = *(const v8h*)&lB[bc * LSTR + klo + 16];
                #pragma unroll
                for (int i = 0; i < 8; ++i) { Bf[ni][i] = b0[i]; Bf[ni][i + 8] = b1[i]; }
            }
            #pragma unroll
            for (int mi = 0; mi < 2; ++mi)
                #pragma unroll
                for (int ni = 0; ni < 2; ++ni)
                    acc[mi][ni] = __builtin_amdgcn_wmma_f32_16x16x32_f16(
                        false, Af[mi], false, Bf[ni], (short)0, acc[mi][ni], false, false);
        }
        __syncthreads();
    }

    // ---- epilogue: bias (+residual) (+ReLU), f16 store ----
    const int mo = (lane >= 16) ? 8 : 0;
    const int l15 = lane & 15;
    const bool hasres = (res != nullptr);
    #pragma unroll
    for (int mi = 0; mi < 2; ++mi) {
        #pragma unroll
        for (int ni = 0; ni < 2; ++ni) {
            const int p  = p0 + wn * 32 + ni * 16 + l15;
            const int oh = p >> woShift;
            const int ow = p & wmask;
            #pragma unroll
            for (int r = 0; r < 8; ++r) {
                const int co = co0 + wm * 32 + mi * 16 + r + mo;
                float v = acc[mi][ni][r] + bias[co];
                size_t off = ((size_t)(n * Cout + co) * Ho + oh) * (size_t)Wo + ow;
                if (hasres) v += (float)res[off];
                if (relu)   v = v > 0.f ? v : 0.f;
                Y[off] = (_Float16)v;
            }
        }
    }
}

// ---------------------------------------------------------------------------
// TDM / async-tensor probe kernel (compiled into the object to exercise the
// CDNA5 tensor-data-mover path; not launched by the pipeline).  Builds a D#
// per CDNA5 ISA: group0 = {count=1, lds_addr, global_addr,
// type=2}; group1 = {data_size=2B, tensor dims/strides, tile dims}.
// ---------------------------------------------------------------------------
#if defined(__HIP_DEVICE_COMPILE__) && __has_builtin(__builtin_amdgcn_tensor_load_to_lds)
__global__ void tdm_probe_kernel(const _Float16* __restrict__ src,
                                 _Float16* __restrict__ dst,
                                 int rows, int cols, int rowStride)
{
    __shared__ _Float16 buf[4096];
    const unsigned lo = (unsigned)(unsigned long long)src;
    const unsigned hi = (unsigned)(((unsigned long long)src) >> 32);

    uint32x4 g0;
    g0[0] = 1u;                                   // count=1, user-mode descriptor
    g0[1] = 0u;                                   // lds_addr = base of LDS alloc
    g0[2] = lo;                                   // global_addr[31:0]
    g0[3] = (hi & 0x01FFFFFFu) | (2u << 30);      // global_addr[56:32] | type=2

    int32x8 g1;
    g1[0] = (1 << 16);                            // wg_mask=0, data_size=1 (2 bytes)
    g1[1] = 0;                                    // no atomic barrier
    g1[2] = cols;                                 // tensor_dim0
    g1[3] = (rows << 16);                         // tensor_dim1 lo | ...
    g1[4] = (64) | (64 << 16);                    // tile_dim0 / tile_dim1
    g1[5] = 0;                                    // tile_dim2 unused
    g1[6] = rowStride;                            // tensor_dim0_stride lo
    g1[7] = 0;                                    // tensor_dim0_stride hi / dim1 stride

    int32x4 g2 = {0, 0, 0, 0};
    int32x4 g3 = {0, 0, 0, 0};

#if __has_include(<hip/amd_detail/amd_gfx1250_TDM.h>)
    int32x8 g4 = {0, 0, 0, 0, 0, 0, 0, 0};       // 6-arg toolchain (therock-10.0)
    __builtin_amdgcn_tensor_load_to_lds(g0, g1, g2, g3, g4, 0);
#else
    __builtin_amdgcn_tensor_load_to_lds(g0, g1, g2, g3, 0);   // ROCm 7.2 form
#endif
    __builtin_amdgcn_s_wait_tensorcnt(0);
    __builtin_amdgcn_s_cluster_barrier();

    const int t = threadIdx.x;
    if (__builtin_amdgcn_cluster_id_x() == 0)
        dst[t] = buf[t];
}
#endif

// ---------------------------------------------------------------------------
// Direct conv (VALU) for irregular layers: Cin=3 input conv, Cout=1, Cin=1,
// and the final conv (+C, clip, f32 out). Tiny share of total FLOPs.
// ---------------------------------------------------------------------------
__global__ void conv_direct_kernel(const _Float16* __restrict__ Xh,
                                   const float*    __restrict__ Xf,
                                   const _Float16* __restrict__ Wt,   // OIHW f16
                                   const float*    __restrict__ bias,
                                   const float*    __restrict__ addf, // optional f32 add
                                   _Float16* Yh, float* Yf,
                                   int Cin, int Cout, int H, int W, int Ho, int Wo,
                                   int ks, int stride, int relu, int clip01)
{
    size_t idx   = (size_t)blockIdx.x * 256 + threadIdx.x;
    size_t total = (size_t)NB * Cout * Ho * Wo;
    if (idx >= total) return;
    int ow = (int)(idx % Wo); size_t t = idx / Wo;
    int oh = (int)(t % Ho);   t /= Ho;
    int co = (int)(t % Cout);
    int n  = (int)(t / Cout);
    int pad = ks >> 1;
    float acc = bias[co];
    for (int ci = 0; ci < Cin; ++ci) {
        for (int ky = 0; ky < ks; ++ky) {
            int ih = oh * stride + ky - pad;
            if ((unsigned)ih >= (unsigned)H) continue;
            size_t xb = ((size_t)(n * Cin + ci) * H + ih) * (size_t)W;
            size_t wb = (((size_t)co * Cin + ci) * ks + ky) * (size_t)ks;
            for (int kx = 0; kx < ks; ++kx) {
                int iw = ow * stride + kx - pad;
                if ((unsigned)iw >= (unsigned)W) continue;
                float xv = Xh ? (float)Xh[xb + iw] : Xf[xb + iw];
                acc += xv * (float)Wt[wb + kx];
            }
        }
    }
    if (relu) acc = fmaxf(acc, 0.f);
    size_t off = ((size_t)(n * Cout + co) * Ho + oh) * (size_t)Wo + ow;
    if (addf)   acc += addf[off];
    if (clip01) acc = fminf(fmaxf(acc, 0.f), 1.f);
    if (Yh) Yh[off] = (_Float16)acc;
    if (Yf) Yf[off] = acc;
}

// ---------------------------------------------------------------------------
// Transposed conv 4x4 stride2 (lhs_dilation=2, pad 2, flipped weights) + ReLU
// + fused skip add.  w layout (Cout, Cin, 4, 4).
// ---------------------------------------------------------------------------
__global__ void upconv_kernel(const _Float16* __restrict__ X,
                              const _Float16* __restrict__ Wt,
                              const float*    __restrict__ bias,
                              const _Float16* __restrict__ skip,
                              _Float16* __restrict__ Y,
                              int Cin, int Cout, int Hi, int Wi)
{
    const int Ho = 2 * Hi, Wo = 2 * Wi;
    size_t idx   = (size_t)blockIdx.x * 256 + threadIdx.x;
    size_t total = (size_t)NB * Cout * Ho * Wo;
    if (idx >= total) return;
    int ow = (int)(idx % Wo); size_t t = idx / Wo;
    int oh = (int)(t % Ho);   t /= Ho;
    int co = (int)(t % Cout);
    int n  = (int)(t / Cout);
    float acc = bias[co];
    for (int ci = 0; ci < Cin; ++ci) {
        size_t xb = (size_t)(n * Cin + ci) * Hi * (size_t)Wi;
        size_t wb = ((size_t)co * Cin + ci) * 16;
        for (int ky = 0; ky < 4; ++ky) {
            int d = oh + ky - 2;
            if (d & 1) continue;
            int ih = d >> 1;
            if ((unsigned)ih >= (unsigned)Hi) continue;
            for (int kx = 0; kx < 4; ++kx) {
                int e = ow + kx - 2;
                if (e & 1) continue;
                int iw = e >> 1;
                if ((unsigned)iw >= (unsigned)Wi) continue;
                acc += (float)X[xb + (size_t)ih * Wi + iw] *
                       (float)Wt[wb + (3 - ky) * 4 + (3 - kx)];  // flipped kernel
            }
        }
    }
    acc = fmaxf(acc, 0.f);
    size_t off = ((size_t)(n * Cout + co) * Ho + oh) * (size_t)Wo + ow;
    acc += (float)skip[off];
    Y[off] = (_Float16)acc;
}

// ---------------------------------------------------------------------------
// diac / sac: separable per-pixel dynamic filtering (edge-clamped 3 taps).
// ---------------------------------------------------------------------------
__global__ void sac_h_kernel(const _Float16* __restrict__ X,
                             const _Float16* __restrict__ F,
                             _Float16* __restrict__ T,
                             int C, int H, int W, int KD, int foff)
{
    size_t idx   = (size_t)blockIdx.x * 256 + threadIdx.x;
    size_t total = (size_t)NB * C * H * W;
    if (idx >= total) return;
    int w = (int)(idx % W); size_t t = idx / W;
    int h = (int)(t % H);   t /= H;
    int c = (int)(t % C);
    int n = (int)(t / C);
    size_t xb = ((size_t)(n * C + c) * H + h) * (size_t)W;
    float acc = 0.f;
    #pragma unroll
    for (int j = 0; j < 3; ++j) {
        int ww = w + j - 1; ww = ww < 0 ? 0 : (ww >= W ? W - 1 : ww);
        size_t fb = (((size_t)n * KD + foff + c * 3 + j) * H + h) * (size_t)W + w;
        acc += (float)X[xb + ww] * (float)F[fb];
    }
    T[xb + w] = (_Float16)acc;
}

__global__ void sac_v_kernel(const _Float16* __restrict__ T,
                             const _Float16* __restrict__ F,
                             _Float16* __restrict__ O,
                             int C, int H, int W, int KD, int foff, int boff, int relu)
{
    size_t idx   = (size_t)blockIdx.x * 256 + threadIdx.x;
    size_t total = (size_t)NB * C * H * W;
    if (idx >= total) return;
    int w = (int)(idx % W); size_t t = idx / W;
    int h = (int)(t % H);   t /= H;
    int c = (int)(t % C);
    int n = (int)(t / C);
    size_t cb = (size_t)(n * C + c) * H * (size_t)W;
    float acc = 0.f;
    #pragma unroll
    for (int j = 0; j < 3; ++j) {
        int hh = h + j - 1; hh = hh < 0 ? 0 : (hh >= H ? H - 1 : hh);
        size_t fb = (((size_t)n * KD + foff + c * 3 + j) * H + h) * (size_t)W + w;
        acc += (float)T[cb + (size_t)hh * W + w] * (float)F[fb];
    }
    acc += (float)F[(((size_t)n * KD + boff + c) * H + h) * (size_t)W + w];
    if (relu) acc = fmaxf(acc, 0.f);
    O[cb + (size_t)h * W + w] = (_Float16)acc;
}

// Channel-offset copy for concatenation.
__global__ void copy_ch_kernel(const _Float16* __restrict__ src,
                               _Float16* __restrict__ dst,
                               int Cs, int Cd, int c0, int HW)
{
    size_t idx   = (size_t)blockIdx.x * 256 + threadIdx.x;
    size_t total = (size_t)NB * Cs * HW;
    if (idx >= total) return;
    int i = (int)(idx % HW); size_t t = idx / HW;
    int c = (int)(t % Cs);
    int n = (int)(t / Cs);
    dst[((size_t)(n * Cd + c0 + c)) * HW + i] = src[idx];
}

// ---------------------------------------------------------------------------
// Host orchestration
// ---------------------------------------------------------------------------
extern "C" void kernel_launch(void* const* d_in, const int* in_sizes, int n_in,
                              void* d_out, int out_size, void* d_ws, size_t ws_size,
                              hipStream_t stream)
{
    (void)in_sizes; (void)n_in; (void)out_size; (void)ws_size;

    // ---- input cursor (setup_inputs dict insertion order, each conv = w,b) ----
    int cur = 0;
    const float* Cimg = (const float*)d_in[cur++];
    struct CP { const float* w; const float* b; int ci, co, ks; };
    auto getc = [&](int ci, int co, int ks) -> CP {
        CP p{ (const float*)d_in[cur], (const float*)d_in[cur + 1], ci, co, ks };
        cur += 2;
        return p;
    };

    static const int ECI[12]  = {3,32,32,32,64,64,64,128,128,128,128,128};
    static const int ECO[12]  = {32,32,32,64,64,64,128,128,128,128,128,128};
    static const int ESTR[12] = {1,1,1,2,1,1,2,1,1,2,1,1};

    CP e[12], k[12];
    for (int i = 0; i < 12; ++i) e[i] = getc(ECI[i], ECO[i], 3);
    for (int i = 0; i < 12; ++i) k[i] = getc(ECI[i], ECO[i], 3);
    CP bpr_c1 = getc(128,128,3);
    CP bpr_r[8];  for (int i = 0; i < 8; ++i) bpr_r[i]  = getc(128,128,3);
    CP bpr_c2 = getc(128,1,3);
    CP conv_bpr = getc(1,128,3);
    CP c44_c1 = getc(256,128,3);
    CP c44_r[8];  for (int i = 0; i < 8; ++i) c44_r[i]  = getc(128,128,3);
    CP c44_c2 = getc(128,128,3);
    CP F_c1 = getc(128,128,3);
    CP F_r[8];    for (int i = 0; i < 8; ++i) F_r[i]    = getc(128,128,3);
    CP F_c2 = getc(128,KDIM,1);
    CP cres_c1 = getc(128,128,3);
    CP cres_r[12]; for (int i = 0; i < 12; ++i) cres_r[i] = getc(128,128,3);
    CP cres_c2 = getc(128,128,3);
    CP up3_u = getc(128,128,4);
    CP up3_r[4]; for (int i = 0; i < 4; ++i) up3_r[i] = getc(128,128,3);
    CP up2_u = getc(128,64,4);
    CP up2_r[4]; for (int i = 0; i < 4; ++i) up2_r[i] = getc(64,64,3);
    CP up1_u = getc(64,32,4);
    CP up1_r[4]; for (int i = 0; i < 4; ++i) up1_r[i] = getc(32,32,3);
    CP outc = getc(32,3,3);

    // ---- workspace bump allocator ----
    char* wsp = (char*)d_ws;
    size_t used = 0;
    auto alloc = [&](size_t bytes) -> void* {
        void* p = wsp + used;
        used += (bytes + 255) & ~(size_t)255;
        return p;
    };

    struct T { _Float16* p; int C, H, W; };

    auto cvtW = [&](CP p) -> _Float16* {
        size_t ne = (size_t)p.co * p.ci * p.ks * p.ks;
        _Float16* o = (_Float16*)alloc(ne * 2);
        wplain_kernel<<<dim3((unsigned)((ne + 255) / 256)), dim3(256), 0, stream>>>(p.w, o, ne);
        return o;
    };

    // WMMA conv: Cin%32==0, Cout%32==0, Wo power of two.
    auto convW = [&](T x, CP p, int stride, bool relu, const _Float16* res) -> T {
        const int Ho = x.H / stride, Wo = x.W / stride;
        size_t wn = (size_t)p.co * p.ci * p.ks * p.ks;
        _Float16* wp = (_Float16*)alloc(wn * 2);
        wpack_kernel<<<dim3((unsigned)((wn + 255) / 256)), dim3(256), 0, stream>>>(
            p.w, wp, p.co, p.ci, p.ks);
        T y{ (_Float16*)alloc((size_t)NB * p.co * Ho * Wo * 2), p.co, Ho, Wo };
        const int WMv = (p.co % 128 == 0) ? 4 : (p.co % 64 == 0) ? 2 : 1;
        const int MT  = WMv * 32;
        const int NT  = (8 / WMv) * 32;
        const int woShift = __builtin_ctz((unsigned)Wo);
        dim3 grid((unsigned)((Ho * Wo) / NT), (unsigned)(p.co / MT), (unsigned)NB);
        if (WMv == 4)
            conv_wmma_kernel<4><<<grid, 256, 0, stream>>>(
                x.p, wp, p.b, res, y.p, p.ci, p.co, x.H, x.W, Ho, Wo, woShift,
                p.ks, stride, relu ? 1 : 0);
        else if (WMv == 2)
            conv_wmma_kernel<2><<<grid, 256, 0, stream>>>(
                x.p, wp, p.b, res, y.p, p.ci, p.co, x.H, x.W, Ho, Wo, woShift,
                p.ks, stride, relu ? 1 : 0);
        else
            conv_wmma_kernel<1><<<grid, 256, 0, stream>>>(
                x.p, wp, p.b, res, y.p, p.ci, p.co, x.H, x.W, Ho, Wo, woShift,
                p.ks, stride, relu ? 1 : 0);
        return y;
    };

    auto resblk = [&](T x, CP a, CP b) -> T {
        T t = convW(x, a, 1, true, nullptr);
        return convW(t, b, 1, false, x.p);   // y = x + conv_b(relu(conv_a(x)))
    };

    // Direct conv for irregular layers.
    auto convD = [&](const _Float16* xh, const float* xf, int H, int W,
                     CP p, int stride, bool relu, const float* addf,
                     bool clip, float* yf) -> T {
        const int Ho = H / stride, Wo = W / stride;
        _Float16* wt = cvtW(p);
        _Float16* yh = nullptr;
        if (!yf) yh = (_Float16*)alloc((size_t)NB * p.co * Ho * Wo * 2);
        size_t total = (size_t)NB * p.co * Ho * Wo;
        conv_direct_kernel<<<dim3((unsigned)((total + 255) / 256)), dim3(256), 0, stream>>>(
            xh, xf, wt, p.b, addf, yh, yf,
            p.ci, p.co, H, W, Ho, Wo, p.ks, stride, relu ? 1 : 0, clip ? 1 : 0);
        return T{ yh, p.co, Ho, Wo };
    };

    // ---- encoders ----
    auto encoder = [&](CP* cv, T* feats) -> T {
        T x = convD(nullptr, Cimg, 256, 256, cv[0], 1, true, nullptr, false, nullptr);
        int fi = 0;
        for (int i = 1; i < 12; ++i) {
            x = convW(x, cv[i], ESTR[i], true, nullptr);
            if (i == 2 || i == 5 || i == 8) feats[fi++] = x;
        }
        return x;
    };

    T ef[3], kf[3];
    T fC = encoder(e, ef);   // features f1,f2,f3 + fC (128 @ 32x32)
    T fk = encoder(k, kf);

    // ---- block4 helper ----
    auto block4 = [&](T x, CP c1, CP* r, CP c2, bool final_act, bool c2wmma) -> T {
        x = convW(x, c1, 1, true, nullptr);
        x = resblk(x, r[0], r[1]);
        x = resblk(x, r[2], r[3]);
        x = resblk(x, r[4], r[5]);
        x = resblk(x, r[6], r[7]);
        if (c2wmma) return convW(x, c2, 1, final_act, nullptr);
        return convD(x.p, nullptr, x.H, x.W, c2, 1, final_act, nullptr, false, nullptr);
    };

    // bpr branch
    T bpr   = block4(fk, bpr_c1, bpr_r, bpr_c2, false, /*c2 Cout=1*/ false);
    T f_bpr = convD(bpr.p, nullptr, bpr.H, bpr.W, conv_bpr, 1, true, nullptr, false, nullptr);

    // concat [fk, f_bpr] -> 256 ch
    const int HW32 = 32 * 32;
    T cat{ (_Float16*)alloc((size_t)NB * 256 * HW32 * 2), 256, 32, 32 };
    {
        size_t tot = (size_t)NB * 128 * HW32;
        copy_ch_kernel<<<dim3((unsigned)((tot + 255) / 256)), dim3(256), 0, stream>>>(
            fk.p, cat.p, 128, 256, 0, HW32);
        copy_ch_kernel<<<dim3((unsigned)((tot + 255) / 256)), dim3(256), 0, stream>>>(
            f_bpr.p, cat.p, 128, 256, 128, HW32);
    }

    T f44  = block4(cat, c44_c1, c44_r, c44_c2, true, true);
    T filt = block4(f44, F_c1, F_r, F_c2, false, true);   // (2, 15232, 32, 32) f16

    // ---- diac: 17 iterations of separable dynamic filtering ----
    _Float16* dtmp = (_Float16*)alloc((size_t)NB * 128 * HW32 * 2);
    _Float16* dbuf[2] = {
        (_Float16*)alloc((size_t)NB * 128 * HW32 * 2),
        (_Float16*)alloc((size_t)NB * 128 * HW32 * 2)
    };
    {
        const size_t tot = (size_t)NB * 128 * HW32;
        const unsigned g = (unsigned)((tot + 255) / 256);
        const _Float16* curp = fC.p;
        for (int i = 0; i < 17; ++i) {
            int f1off = i * 768;
            int f2off = i * 768 + 384;
            int boff  = 17 * 768 + i * 128;
            _Float16* op = dbuf[i & 1];
            sac_h_kernel<<<g, 256, 0, stream>>>(curp, filt.p, dtmp, 128, 32, 32, KDIM, f1off);
            sac_v_kernel<<<g, 256, 0, stream>>>(dtmp, filt.p, op, 128, 32, 32, KDIM,
                                                f2off, boff, (i < 16) ? 1 : 0);
            curp = op;
        }
        fC.p = (_Float16*)curp;   // diac output (2,128,32,32)
    }

    // ---- cres ----
    T x = convW(fC, cres_c1, 1, true, nullptr);
    x = resblk(x, cres_r[0], cres_r[1]);
    x = resblk(x, cres_r[2], cres_r[3]);
    x = resblk(x, cres_r[4], cres_r[5]);
    x = convW(x, cres_c2, 1, true, nullptr);

    // ---- decoder: upconv + skip + resblocks ----
    auto upconv = [&](T xin, CP u, T skip) -> T {
        const int Ho = xin.H * 2, Wo = xin.W * 2;
        _Float16* wt = cvtW(u);
        T y{ (_Float16*)alloc((size_t)NB * u.co * Ho * Wo * 2), u.co, Ho, Wo };
        size_t tot = (size_t)NB * u.co * Ho * Wo;
        upconv_kernel<<<dim3((unsigned)((tot + 255) / 256)), dim3(256), 0, stream>>>(
            xin.p, wt, u.b, skip.p, y.p, u.ci, u.co, xin.H, xin.W);
        return y;
    };

    x = upconv(x, up3_u, ef[2]);
    x = resblk(x, up3_r[0], up3_r[1]);
    x = resblk(x, up3_r[2], up3_r[3]);

    x = upconv(x, up2_u, ef[1]);
    x = resblk(x, up2_r[0], up2_r[1]);
    x = resblk(x, up2_r[2], up2_r[3]);

    x = upconv(x, up1_u, ef[0]);
    x = resblk(x, up1_r[0], up1_r[1]);
    x = resblk(x, up1_r[2], up1_r[3]);

    // ---- final conv + ReLU + add C + clip -> f32 d_out ----
    convD(x.p, nullptr, 256, 256, outc, 1, true, Cimg, true, (float*)d_out);
}
